// DeeperGCN_70695161692371
// MI455X (gfx1250) — matrix operand order repaired and checked by reading, not compiled, not run on hardware
//
#include <hip/hip_runtime.h>
#include <hip/hip_bf16.h>
#include <math.h>

// ---------------------------------------------------------------------------
// DeeperGCN forward for MI455X (gfx1250, wave32).
// Per-edge MLP (192->128->64), node encoder and root weights run on
// v_wmma_f32_16x16x32_bf16 (fp32 accumulate). Edge-MLP weights are staged in
// LDS once per block (async load-to-LDS, ASYNCcnt tracked) and each wave
// grid-strides over 16-edge tiles. Scatter-softmax = 2-pass max + num/den
// atomics resident in L2 (12.8 MB buffers << 192 MB L2).
// ---------------------------------------------------------------------------

typedef __attribute__((ext_vector_type(16))) __bf16 v16bf;
typedef __attribute__((ext_vector_type(8)))  float  v8f;
typedef int vint4 __attribute__((vector_size(16)));

#define AS1 __attribute__((address_space(1)))
#define AS3 __attribute__((address_space(3)))

union V16 { uint4 u[2]; v16bf v; unsigned short s[16]; __bf16 h[16]; };

__device__ inline unsigned short f2bf(float f) {
    __bf16 b = (__bf16)f;                       // native v_cvt on gfx1250
    union { __bf16 b; unsigned short s; } w; w.b = b;
    return w.s;
}

// A-fragment (16-bit, 16x32): lane holds row lane&15; elements 0..7 at p[0..7],
// 8..15 at p[16..23]; caller pre-adds kb + 8*(lane>>4).
__device__ inline v16bf load_a_frag(const unsigned short* p) {
    V16 t;
    t.u[0] = *(const uint4*)(p);
    t.u[1] = *(const uint4*)(p + 16);
    return t.v;
}
// B-fragment (16-bit, 32x16) from column-major weights: elements 0..7 at p[0..7],
// 8..15 at p[16..23]; caller pre-adds kb + 8*(lane>>4).
__device__ inline v16bf load_b_frag(const unsigned short* p) {
    V16 t;
    t.u[0] = *(const uint4*)(p);
    t.u[1] = *(const uint4*)(p + 16);
    return t.v;
}
// A-fragment built from fp32 source rows (node encoder), converted on the fly.
__device__ inline v16bf load_a_frag_f32(const float* p) {
    V16 t;
#pragma unroll
    for (int i = 0; i < 8; ++i) t.h[i]     = (__bf16)p[i];
#pragma unroll
    for (int i = 0; i < 8; ++i) t.h[8 + i] = (__bf16)p[16 + i];
    return t.v;
}
__device__ inline v8f wmma_bf16(v16bf a, v16bf b, v8f c) {
    return __builtin_amdgcn_wmma_f32_16x16x32_bf16(false, a, false, b, (short)0, c, false, false);
}
__device__ inline void atomicMaxF(float* addr, float v) {
    if (v >= 0.f) atomicMax((int*)addr, __float_as_int(v));
    else          atomicMin((unsigned int*)addr, __float_as_uint(v));
}
__device__ inline v8f v8f_zero() { return (v8f){0.f,0.f,0.f,0.f,0.f,0.f,0.f,0.f}; }

// 16-byte global -> LDS copy; async (ASYNCcnt) path on gfx1250 toolchains that
// expose the builtin, plain load/store otherwise.
__device__ inline void copy16_to_lds(const void* g, void* l) {
#if __has_builtin(__builtin_amdgcn_global_load_async_to_lds_b128)
    __builtin_amdgcn_global_load_async_to_lds_b128(
        (AS1 vint4*)(g), (AS3 vint4*)(l), 0, 0);
#else
    *(uint4*)l = *(const uint4*)g;
#endif
}
__device__ inline void copy_to_lds_wait() {
#if __has_builtin(__builtin_amdgcn_s_wait_asynccnt)
    __builtin_amdgcn_s_wait_asynccnt(0);
#endif
}

// --------------------------- weight prep -----------------------------------
// src [K][Ncol] f32 row-major  ->  dst [Ncol][K] bf16 (column-major weights)
__global__ __launch_bounds__(256) void transpose_bf16_kernel(
    const float* __restrict__ src, unsigned short* __restrict__ dst, int K, int Ncol) {
    int i = blockIdx.x * 256 + threadIdx.x;
    if (i >= K * Ncol) return;
    int k = i / Ncol, n = i - k * Ncol;
    dst[(size_t)n * K + k] = f2bf(src[i]);
}

// --------------------------- node encoder ----------------------------------
// h = x @ node_W + node_b ; also emits hb = bf16(h) (layer-0 conv input)
__global__ __launch_bounds__(128) void node_encoder_kernel(
    const float* __restrict__ x, const unsigned short* __restrict__ WT,
    const float* __restrict__ bias, float* __restrict__ h,
    unsigned short* __restrict__ hb, int Nn) {
    int wave = threadIdx.x >> 5, lane = threadIdx.x & 31;
    int tile = blockIdx.x * 4 + wave;
    if (tile * 16 >= Nn) return;
    int nbase = tile * 16;
    int row = lane & 15;
    const float* rowp = x + (size_t)(nbase + row) * 128;
    int aoff = (lane >> 4) << 3;     // +0 / +8
    int boff = (lane >> 4) << 3;     // +0 / +8
    int ncol = lane & 15;
    v8f acc[4];
#pragma unroll
    for (int n = 0; n < 4; ++n) acc[n] = v8f_zero();
#pragma unroll
    for (int ks = 0; ks < 4; ++ks) {
        int kb = ks * 32;
        v16bf a = load_a_frag_f32(rowp + kb + aoff);
#pragma unroll
        for (int n = 0; n < 4; ++n) {
            v16bf b = load_b_frag(WT + (size_t)(n * 16 + ncol) * 128 + kb + boff);
            acc[n] = wmma_bf16(a, b, acc[n]);
        }
    }
    int mrow_add = (lane >> 4) << 3;
#pragma unroll
    for (int n = 0; n < 4; ++n) {
        int col = n * 16 + ncol;
        float bv = bias[col];
#pragma unroll
        for (int r = 0; r < 8; ++r) {
            int node = nbase + r + mrow_add;
            float v = acc[n][r] + bv;
            h[(size_t)node * 64 + col]  = v;
            hb[(size_t)node * 64 + col] = f2bf(v);
        }
    }
}

// --------------------------- edge encoder ----------------------------------
// e = edge_attr @ edge_W + edge_b (K=16: bandwidth bound)
__global__ __launch_bounds__(256) void edge_enc_kernel(
    const float* __restrict__ ea, const float* __restrict__ W,
    const float* __restrict__ bias, float* __restrict__ e_out,
    unsigned short* __restrict__ eb_out, int En) {
    __shared__ float Ws[16 * 64];
    __shared__ float bs[64];
    for (int i = threadIdx.x; i < 16 * 64; i += 256) Ws[i] = W[i];
    for (int i = threadIdx.x; i < 64; i += 256) bs[i] = bias[i];
    __syncthreads();
    int e = blockIdx.x * 256 + threadIdx.x;
    if (e >= En) return;
    float a[16];
#pragma unroll
    for (int k = 0; k < 16; ++k) a[k] = ea[(size_t)e * 16 + k];
    for (int c = 0; c < 64; ++c) {
        float s = bs[c];
#pragma unroll
        for (int k = 0; k < 16; ++k) s += a[k] * Ws[k * 64 + c];
        e_out[(size_t)e * 64 + c]  = s;
        eb_out[(size_t)e * 64 + c] = f2bf(s);
    }
}

// ------------------------- per-row layernorm -> bf16 ------------------------
__global__ __launch_bounds__(256) void ln_bf16_kernel(
    const float* __restrict__ src, const float* __restrict__ g,
    const float* __restrict__ b, unsigned short* __restrict__ dst,
    int rows, int do_relu) {
    int i = blockIdx.x * 256 + threadIdx.x;
    if (i >= rows) return;
    const float* r = src + (size_t)i * 64;
    float mu = 0.f;
#pragma unroll
    for (int c = 0; c < 64; ++c) mu += r[c];
    mu *= (1.f / 64.f);
    float var = 0.f;
#pragma unroll
    for (int c = 0; c < 64; ++c) { float d = r[c] - mu; var += d * d; }
    var *= (1.f / 64.f);
    float inv = rsqrtf(var + 1e-5f);
#pragma unroll
    for (int c = 0; c < 64; ++c) {
        float v = (r[c] - mu) * inv * g[c] + b[c];
        if (do_relu) v = fmaxf(v, 0.f);
        dst[(size_t)i * 64 + c] = f2bf(v);
    }
}

// --------------------------- segment buffers init ---------------------------
__global__ __launch_bounds__(256) void fill3_kernel(
    float* __restrict__ smax, float* __restrict__ num, float* __restrict__ den, int total) {
    int i = blockIdx.x * 256 + threadIdx.x;
    if (i >= total) return;
    smax[i] = -3.402823466e38f;
    num[i] = 0.f;
    den[i] = 0.f;
}

// --------------------------- edge MLP (the hot loop) ------------------------
// m = relu(cat(h[dst],h[src],e) @ W1 + b1) @ W2 + b2 ; atomicMax(m*t) into smax
// Weights staged in LDS once per block; waves grid-stride over 16-edge tiles.
// LDS row pads (200 / 136 shorts) keep 16-lane ds_load_b128 conflict-free.
__global__ __launch_bounds__(128) void edge_mlp_kernel(
    const unsigned short* __restrict__ hb, const unsigned short* __restrict__ eb,
    const long long* __restrict__ srcI, const long long* __restrict__ dstI,
    const unsigned short* __restrict__ W1T, const float* __restrict__ b1,
    const unsigned short* __restrict__ W2T, const float* __restrict__ b2,
    const float* __restrict__ tptr, int layer,
    float* __restrict__ m_out, float* __restrict__ smax, int En) {
    __shared__ __align__(16) unsigned short W1s[128][200];  // 192 used, pad 200
    __shared__ __align__(16) unsigned short W2s[64][136];   // 128 used, pad 136
    __shared__ __align__(16) unsigned short hid[4][16][136];
    __shared__ float b1s[128];
    __shared__ float b2s[64];
    __shared__ int sdst[4][16];

    // ---- cooperative weight staging (async -> LDS, ASYNCcnt tracked)
    for (int i = threadIdx.x; i < 128 * 24; i += 128) {       // 24 x 16B per row
        int r = i / 24, q = i - r * 24;
        copy16_to_lds(W1T + (size_t)r * 192 + q * 8, &W1s[r][q * 8]);
    }
    for (int i = threadIdx.x; i < 64 * 16; i += 128) {        // 16 x 16B per row
        int r = i / 16, q = i - r * 16;
        copy16_to_lds(W2T + (size_t)r * 128 + q * 8, &W2s[r][q * 8]);
    }
    if (threadIdx.x < 128) b1s[threadIdx.x] = b1[threadIdx.x];
    if (threadIdx.x < 64)  b2s[threadIdx.x] = b2[threadIdx.x];
    copy_to_lds_wait();
    __syncthreads();

    int wave = threadIdx.x >> 5, lane = threadIdx.x & 31;
    float t = tptr[layer];
    int aoff = (lane >> 4) << 3;
    int boff = (lane >> 4) << 3;
    int ncol = lane & 15;
    int mrow_add = (lane >> 4) << 3;
    int row = lane & 15;
    int ntiles = En >> 4;
    int stride = gridDim.x * 4;

    for (int tile = blockIdx.x * 4 + wave; tile < ntiles; tile += stride) {
        int ebase = tile * 16;
        int myedge = ebase + row;
        int sidx = (int)srcI[myedge];
        int didx = (int)dstI[myedge];
        if (lane < 16) sdst[wave][lane] = didx;

        const unsigned short* rp_d = hb + (size_t)didx * 64;
        const unsigned short* rp_s = hb + (size_t)sidx * 64;
        const unsigned short* rp_e = eb + (size_t)myedge * 64;

        // GEMM1: [16,192] x [192,128]
        v8f acc1[8];
#pragma unroll
        for (int n = 0; n < 8; ++n) acc1[n] = v8f_zero();
#pragma unroll
        for (int ks = 0; ks < 6; ++ks) {
            int kb = ks * 32;
            const unsigned short* rp;
            int koff;
            if (kb < 64)       { rp = rp_d; koff = kb;       }
            else if (kb < 128) { rp = rp_s; koff = kb - 64;  }
            else               { rp = rp_e; koff = kb - 128; }
            v16bf a = load_a_frag(rp + koff + aoff);
#pragma unroll
            for (int n = 0; n < 8; ++n) {
                v16bf b = load_b_frag(&W1s[n * 16 + ncol][kb + boff]);
                acc1[n] = wmma_bf16(a, b, acc1[n]);
            }
        }
        // bias + ReLU -> LDS bf16 (re-layout for GEMM2 A-fragments)
#pragma unroll
        for (int n = 0; n < 8; ++n) {
            int col = n * 16 + ncol;
            float bv = b1s[col];
#pragma unroll
            for (int r = 0; r < 8; ++r) {
                float v = acc1[n][r] + bv;
                hid[wave][r + mrow_add][col] = f2bf(fmaxf(v, 0.f));
            }
        }
        // GEMM2: [16,128] x [128,64]  (A from LDS; wave-private, DS in-order)
        v8f acc2[4];
#pragma unroll
        for (int n = 0; n < 4; ++n) acc2[n] = v8f_zero();
        const unsigned short* hrow = &hid[wave][lane & 15][0];
#pragma unroll
        for (int ks = 0; ks < 4; ++ks) {
            int kb = ks * 32;
            v16bf a = load_a_frag(hrow + kb + aoff);
#pragma unroll
            for (int n = 0; n < 4; ++n) {
                v16bf b = load_b_frag(&W2s[n * 16 + ncol][kb + boff]);
                acc2[n] = wmma_bf16(a, b, acc2[n]);
            }
        }
        // bias, store m, scatter-max of logits
#pragma unroll
        for (int n = 0; n < 4; ++n) {
            int col = n * 16 + ncol;
            float bv = b2s[col];
#pragma unroll
            for (int r = 0; r < 8; ++r) {
                int mr = r + mrow_add;
                float mv = acc2[n][r] + bv;
                m_out[(size_t)(ebase + mr) * 64 + col] = mv;
                atomicMaxF(&smax[(size_t)sdst[wave][mr] * 64 + col], mv * t);
            }
        }
    }
}

// ------------------- scatter pass: num/den accumulation ---------------------
// out = segsum(m*exp(l-max)) / segsum(exp(l-max)); also e += m (or e = m).
// One thread per (edge, 8-channel group): one dst load per 8 channels.
__global__ __launch_bounds__(256) void scatter_kernel(
    const float* __restrict__ m, const long long* __restrict__ dstI,
    const float* __restrict__ smax, float* __restrict__ num, float* __restrict__ den,
    float* __restrict__ e_io, const float* __restrict__ tptr, int layer,
    int first, int En) {
    int i = blockIdx.x * 256 + threadIdx.x;
    if (i >= En * 8) return;
    int e = i >> 3, g = i & 7;
    int c0 = g * 8;
    float t = tptr[layer];
    int d = (int)dstI[e];
    const float* mp = m    + (size_t)e * 64 + c0;
    float*       ep = e_io + (size_t)e * 64 + c0;
    const float* sp = smax + (size_t)d * 64 + c0;
    float*       np = num  + (size_t)d * 64 + c0;
    float*       dp = den  + (size_t)d * 64 + c0;
#pragma unroll
    for (int c = 0; c < 8; ++c) {
        float mv = mp[c];
        float p = __expf(mv * t - sp[c]);
        atomicAdd(&np[c], mv * p);
        atomicAdd(&dp[c], p);
        ep[c] = first ? mv : (ep[c] + mv);
    }
}

// --------------------------- node update ------------------------------------
// dh = num/den + hn @ Wr ; h = first ? dh : h + dh
__global__ __launch_bounds__(128) void node_update_kernel(
    const unsigned short* __restrict__ hb, const unsigned short* __restrict__ WrT,
    const float* __restrict__ num, const float* __restrict__ den,
    float* __restrict__ h, int first, int Nn) {
    int wave = threadIdx.x >> 5, lane = threadIdx.x & 31;
    int tile = blockIdx.x * 4 + wave;
    if (tile * 16 >= Nn) return;
    int nbase = tile * 16;
    int row = lane & 15;
    const unsigned short* rp = hb + (size_t)(nbase + row) * 64;
    int aoff = (lane >> 4) << 3;
    int boff = (lane >> 4) << 3;
    int ncol = lane & 15;
    v8f acc[4];
#pragma unroll
    for (int n = 0; n < 4; ++n) acc[n] = v8f_zero();
#pragma unroll
    for (int ks = 0; ks < 2; ++ks) {
        int kb = ks * 32;
        v16bf a = load_a_frag(rp + kb + aoff);
#pragma unroll
        for (int n = 0; n < 4; ++n) {
            v16bf b = load_b_frag(WrT + (size_t)(n * 16 + ncol) * 64 + kb + boff);
            acc[n] = wmma_bf16(a, b, acc[n]);
        }
    }
    int mrow_add = (lane >> 4) << 3;
#pragma unroll
    for (int n = 0; n < 4; ++n) {
        int col = n * 16 + ncol;
#pragma unroll
        for (int r = 0; r < 8; ++r) {
            int node = nbase + r + mrow_add;
            size_t ix = (size_t)node * 64 + col;
            float dv = den[ix];
            float agg = (dv > 0.f) ? (num[ix] / dv) : 0.f;
            float out = agg + acc[n][r];
            h[ix] = first ? out : (h[ix] + out);
        }
    }
}

// --------------------------- final LN + ReLU --------------------------------
__global__ __launch_bounds__(256) void final_ln_kernel(
    const float* __restrict__ h, const float* __restrict__ g,
    const float* __restrict__ b, float* __restrict__ out, int rows) {
    int i = blockIdx.x * 256 + threadIdx.x;
    if (i >= rows) return;
    const float* r = h + (size_t)i * 64;
    float mu = 0.f;
#pragma unroll
    for (int c = 0; c < 64; ++c) mu += r[c];
    mu *= (1.f / 64.f);
    float var = 0.f;
#pragma unroll
    for (int c = 0; c < 64; ++c) { float d = r[c] - mu; var += d * d; }
    var *= (1.f / 64.f);
    float inv = rsqrtf(var + 1e-5f);
#pragma unroll
    for (int c = 0; c < 64; ++c) {
        float v = (r[c] - mu) * inv * g[c] + b[c];
        out[(size_t)i * 64 + c] = fmaxf(v, 0.f);
    }
}

// ============================================================================
extern "C" void kernel_launch(void* const* d_in, const int* in_sizes, int n_in,
                              void* d_out, int out_size, void* d_ws, size_t ws_size,
                              hipStream_t stream) {
    const float*     x       = (const float*)d_in[0];
    const long long* eidx    = (const long long*)d_in[1];  // jnp.int64
    const float*     eattr   = (const float*)d_in[2];
    const float*     node_W  = (const float*)d_in[3];
    const float*     node_b  = (const float*)d_in[4];
    const float*     edge_W  = (const float*)d_in[5];
    const float*     edge_b  = (const float*)d_in[6];
    const float*     conv_W1 = (const float*)d_in[7];
    const float*     conv_b1 = (const float*)d_in[8];
    const float*     conv_W2 = (const float*)d_in[9];
    const float*     conv_b2 = (const float*)d_in[10];
    const float*     conv_Wr = (const float*)d_in[11];
    const float*     conv_t  = (const float*)d_in[12];
    const float*     ln_g    = (const float*)d_in[13];
    const float*     ln_b    = (const float*)d_in[14];
    const float*     eln_g   = (const float*)d_in[15];
    const float*     eln_b   = (const float*)d_in[16];
    (void)n_in; (void)out_size; (void)ws_size;

    const int N = in_sizes[0] / 128;
    const int E = in_sizes[1] / 2;
    const long long* srcI = eidx;
    const long long* dstI = eidx + E;

    char* ws = (char*)d_ws;
    size_t off = 0;
    auto take = [&](size_t bytes) -> char* {
        off = (off + 255) & ~(size_t)255;
        char* p = ws + off;
        off += bytes;
        return p;
    };
    float*          h      = (float*)take((size_t)N * 64 * 4);
    unsigned short* hb     = (unsigned short*)take((size_t)N * 64 * 2);
    float*          e      = (float*)take((size_t)E * 64 * 4);
    unsigned short* eb     = (unsigned short*)take((size_t)E * 64 * 2);
    float*          m      = (float*)take((size_t)E * 64 * 4);
    float*          smax   = (float*)take((size_t)N * 64 * 4);
    float*          num    = (float*)take((size_t)N * 64 * 4);
    float*          den    = (float*)take((size_t)N * 64 * 4);
    unsigned short* nodeWT = (unsigned short*)take((size_t)64 * 128 * 2);
    unsigned short* W1T    = (unsigned short*)take((size_t)3 * 128 * 192 * 2);
    unsigned short* W2T    = (unsigned short*)take((size_t)3 * 64 * 128 * 2);
    unsigned short* WrT    = (unsigned short*)take((size_t)3 * 64 * 64 * 2);

    auto cdiv = [](int a, int b) { return (a + b - 1) / b; };

    // ---- weight prep: f32 row-major -> bf16 column-major
    transpose_bf16_kernel<<<cdiv(128 * 64, 256), 256, 0, stream>>>(node_W, nodeWT, 128, 64);
    for (int l = 0; l < 3; ++l) {
        transpose_bf16_kernel<<<cdiv(192 * 128, 256), 256, 0, stream>>>(
            conv_W1 + (size_t)l * 192 * 128, W1T + (size_t)l * 128 * 192, 192, 128);
        transpose_bf16_kernel<<<cdiv(128 * 64, 256), 256, 0, stream>>>(
            conv_W2 + (size_t)l * 128 * 64, W2T + (size_t)l * 64 * 128, 128, 64);
        transpose_bf16_kernel<<<cdiv(64 * 64, 256), 256, 0, stream>>>(
            conv_Wr + (size_t)l * 64 * 64, WrT + (size_t)l * 64 * 64, 64, 64);
    }

    // ---- encoders
    node_encoder_kernel<<<cdiv(N / 16, 4), 128, 0, stream>>>(x, nodeWT, node_b, h, hb, N);
    edge_enc_kernel<<<cdiv(E, 256), 256, 0, stream>>>(eattr, edge_W, edge_b, e, eb, E);

    // ---- 3 SAGE convs (layer 0 bare, 1..2 res+)
    const int mlp_blocks = 512;   // persistent-ish: amortize 64KB LDS weight stage
    for (int l = 0; l < 3; ++l) {
        if (l > 0) {
            ln_bf16_kernel<<<cdiv(N, 256), 256, 0, stream>>>(
                h, ln_g + l * 64, ln_b + l * 64, hb, N, /*relu=*/1);
            ln_bf16_kernel<<<cdiv(E, 256), 256, 0, stream>>>(
                e, eln_g + (l - 1) * 64, eln_b + (l - 1) * 64, eb, E, /*relu=*/0);
        }
        fill3_kernel<<<cdiv(N * 64, 256), 256, 0, stream>>>(smax, num, den, N * 64);
        edge_mlp_kernel<<<mlp_blocks, 128, 0, stream>>>(
            hb, eb, srcI, dstI,
            W1T + (size_t)l * 128 * 192, conv_b1 + l * 128,
            W2T + (size_t)l * 64 * 128, conv_b2 + l * 64,
            conv_t, l, m, smax, E);
        scatter_kernel<<<cdiv(E * 8, 256), 256, 0, stream>>>(
            m, dstI, smax, num, den, e, conv_t, l, (l == 0) ? 1 : 0, E);
        node_update_kernel<<<cdiv(N / 16, 4), 128, 0, stream>>>(
            hb, WrT + (size_t)l * 64 * 64, num, den, h, (l == 0) ? 1 : 0, N);
    }

    // ---- final norm + relu (layer-0 LN params)
    final_ln_kernel<<<cdiv(N, 256), 256, 0, stream>>>(h, ln_g, ln_b, (float*)d_out, N);
}
